// CloudNet_69939247448465
// MI455X (gfx1250) — compile-verified, matrix operand order checked by reading, not run on hardware
//
#include <hip/hip_runtime.h>
#include <hip/hip_bf16.h>
#include <math.h>

typedef __attribute__((ext_vector_type(16))) _Float16 v16h;
typedef __attribute__((ext_vector_type(8)))  _Float16 v8h;
typedef __attribute__((ext_vector_type(8)))  float    v8f;

#define NEGBIG (-1e30f)
#define FBIG   (3.4e38f)

// ---------------------------------------------------------------------------
// WMMA helpers (CDNA5 gfx1250, wave32): D = A(16x32 f16) x B(32x16 f16) + C(f32)
// ---------------------------------------------------------------------------
__device__ __forceinline__ v8f wmma_f16(v16h a, v16h b, v8f c) {
  return __builtin_amdgcn_wmma_f32_16x16x32_f16(
      /*neg_a=*/false, a, /*neg_b=*/false, b,
      /*c_mod=*/(short)0, c, /*reuse_a=*/false, /*reuse_b=*/false);
}

// A-fragment from row-major LDS [rows][ldk] f16: rows row0..+15, K cols k0..+31.
// ISA 7.12.2: lane m=L%16, khi=(L/16)*8; per lane the fragment is two contiguous
// 8-half runs: K = khi..khi+7 and K = khi+16..khi+23  ->  2x ds_load_b128.
__device__ __forceinline__ v16h load_a_frag(const _Float16* buf, int row0, int ldk, int k0) {
  int lane = threadIdx.x & 31;
  int m    = lane & 15;
  int khi  = (lane >> 4) * 8;
  const _Float16* r = buf + (size_t)(row0 + m) * ldk + k0 + khi;
  v8h lo = *(const v8h*)(r);
  v8h hi = *(const v8h*)(r + 16);
  return __builtin_shufflevector(lo, hi, 0, 1, 2, 3, 4, 5, 6, 7,
                                 8, 9, 10, 11, 12, 13, 14, 15);
}

// B-fragment from PRE-SWIZZLED weights: layout [kchunk][ntile][lane][16] halfs,
// i.e. each lane's 16 halfs are contiguous -> one 32B load, no packing ALU.
__device__ __forceinline__ v16h load_b_frag(const _Float16* w, int H, int k0, int n0) {
  int lane = threadIdx.x & 31;
  size_t tile = (size_t)((k0 >> 5) * (H >> 4) + (n0 >> 4));
  return *(const v16h*)(w + (tile * 32 + lane) * 16);
}

// Store a 16x16 f32 accumulator tile (bias + optional relu) as f16 row-major.
// C/D layout: lane L, vgpr v -> M = v + 8*(L/16), N = L%16.
__device__ __forceinline__ void store_d_f16(v8f acc, const float* bias, int n0,
                                            _Float16* out, int row0, int ldn, bool relu) {
  int lane = threadIdx.x & 31;
  int n    = lane & 15;
  int mhi  = (lane >> 4) * 8;
  float bv = bias[n0 + n];
#pragma unroll
  for (int v = 0; v < 8; ++v) {
    float x = acc[v] + bv;
    if (relu) x = fmaxf(x, 0.0f);
    out[(size_t)(row0 + mhi + v) * ldn + n0 + n] = (_Float16)x;
  }
}

// One wave pushes its 16-row tile through a dense layer: in[rows][Kp] -> out[rows][H]
__device__ __forceinline__ void wave_layer(const _Float16* in, int Kp,
                                           const _Float16* w, const float* bias, int H,
                                           _Float16* out, int row0, bool relu) {
  for (int n0 = 0; n0 < H; n0 += 16) {
    v8f acc = {};
#pragma unroll 2
    for (int k0 = 0; k0 < Kp; k0 += 32) {
      v16h a = load_a_frag(in, row0, Kp, k0);
      v16h b = load_b_frag(w, H, k0, n0);
      acc = wmma_f16(a, b, acc);
    }
    store_d_f16(acc, bias, n0, out, row0, H, relu);
  }
}

// ---------------------------------------------------------------------------
// Weight convert + swizzle: f32 [K][N] -> f16 fragment order [Kp/32][N/16][32][16]
// (zero-filled for K..Kp-1). Output element i:
//   e=i&15, lane=(i>>4)&31, tile=i>>9, kc=tile/(N/16), nt=tile%(N/16)
//   g=e>>1; k = kc*32 + (lane>>4)*8 + (g&3)*2 + (g>>2)*16 + (e&1); n = nt*16+(lane&15)
// ---------------------------------------------------------------------------
__global__ void cvt_w_kernel(const float* __restrict__ w, _Float16* __restrict__ o,
                             int K, int N, int Kp) {
  int i = blockIdx.x * 256 + threadIdx.x;
  if (i >= Kp * N) return;
  int e    = i & 15;
  int lane = (i >> 4) & 31;
  int tile = i >> 9;
  int NT   = N >> 4;
  int kc = tile / NT, nt = tile - kc * NT;
  int g = e >> 1;
  int k = kc * 32 + ((lane >> 4) * 8) + ((g & 3) * 2) + ((g >> 2) * 16) + (e & 1);
  int n = nt * 16 + (lane & 15);
  o[i] = (k < K) ? (_Float16)w[(size_t)k * N + n] : (_Float16)0.0f;
}

// ---------------------------------------------------------------------------
// Farthest point sampling: one block per cloud, deterministic start at 0
// ---------------------------------------------------------------------------
__global__ void fps_kernel(const float* __restrict__ pos, int N, int M,
                           int* __restrict__ idx_out) {
  __shared__ float mind[4096];
  __shared__ float smax[256];
  __shared__ int   simax[256];
  int b = blockIdx.x, tid = threadIdx.x;
  const float* p = pos + (size_t)b * N * 3;
  for (int i = tid; i < N; i += 256) mind[i] = FBIG;
  if (tid == 0) idx_out[(size_t)b * M] = 0;
  int last = 0;
  __syncthreads();
  for (int s = 1; s < M; ++s) {
    float lx = p[last * 3], ly = p[last * 3 + 1], lz = p[last * 3 + 2];
    float best = -1.0f; int bi = 0;
    for (int i = tid; i < N; i += 256) {
      float dx = p[i * 3] - lx, dy = p[i * 3 + 1] - ly, dz = p[i * 3 + 2] - lz;
      float m = fminf(mind[i], dx * dx + dy * dy + dz * dz);
      mind[i] = m;
      if (m > best) { best = m; bi = i; }
    }
    smax[tid] = best; simax[tid] = bi;
    __syncthreads();
    for (int off = 128; off > 0; off >>= 1) {
      if (tid < off && smax[tid + off] > smax[tid]) {
        smax[tid] = smax[tid + off]; simax[tid] = simax[tid + off];
      }
      __syncthreads();
    }
    last = simax[0];
    if (tid == 0) idx_out[(size_t)b * M + s] = last;
    __syncthreads();
  }
}

__global__ void gather_pos_kernel(const float* __restrict__ pos, const int* __restrict__ idx,
                                  float* __restrict__ out, int N, int M, int B) {
  int i = blockIdx.x * 256 + threadIdx.x;
  if (i >= B * M * 3) return;
  int c = i % 3, r = i / 3, m = r % M, b = r / M;
  int j = idx[(size_t)b * M + m];
  out[i] = pos[((size_t)b * N + j) * 3 + c];
}

// ---------------------------------------------------------------------------
// Radius k-NN: one block per query; k=64 nearest within r (sorted ascending)
// ---------------------------------------------------------------------------
__global__ void radius_kernel(const float* __restrict__ pos_all, const float* __restrict__ pos_q,
                              int N, int M, float r2,
                              int* __restrict__ nidx, unsigned char* __restrict__ nvalid) {
  __shared__ float d2s[4096];
  __shared__ float smin[256];
  __shared__ int   simin[256];
  int q = blockIdx.x, tid = threadIdx.x;
  int b = q / M;
  const float* p = pos_all + (size_t)b * N * 3;
  float qx = pos_q[(size_t)q * 3], qy = pos_q[(size_t)q * 3 + 1], qz = pos_q[(size_t)q * 3 + 2];
  for (int i = tid; i < N; i += 256) {
    float dx = p[i * 3] - qx, dy = p[i * 3 + 1] - qy, dz = p[i * 3 + 2] - qz;
    d2s[i] = dx * dx + dy * dy + dz * dz;
  }
  __syncthreads();
  for (int k = 0; k < 64; ++k) {
    float best = FBIG; int bi = 0;
    for (int i = tid; i < N; i += 256) {
      float d = d2s[i];
      if (d < best) { best = d; bi = i; }
    }
    smin[tid] = best; simin[tid] = bi;
    __syncthreads();
    for (int off = 128; off > 0; off >>= 1) {
      if (tid < off && smin[tid + off] < smin[tid]) {
        smin[tid] = smin[tid + off]; simin[tid] = simin[tid + off];
      }
      __syncthreads();
    }
    if (tid == 0) {
      int sel = simin[0];
      bool ok = (smin[0] <= r2);
      nidx[(size_t)q * 64 + k]   = ok ? sel : 0;
      nvalid[(size_t)q * 64 + k] = ok ? 1 : 0;
      d2s[sel] = FBIG;  // exclude from next selection
    }
    __syncthreads();
  }
}

// ---------------------------------------------------------------------------
// PointConv: block = 1 query (64 neighbors), 4 waves x 16-row tiles, WMMA MLP,
// masked max aggregation + fused ReLU
// ---------------------------------------------------------------------------
__global__ void point_conv_kernel(const float* __restrict__ pos_src, const float* __restrict__ pos_dst,
                                  const float* __restrict__ x_src, const int* __restrict__ nidx,
                                  const unsigned char* __restrict__ nvalid,
                                  const _Float16* __restrict__ w1, const float* __restrict__ b1,
                                  const _Float16* __restrict__ w2, const float* __restrict__ b2,
                                  const _Float16* __restrict__ w3, const float* __restrict__ b3,
                                  float* __restrict__ x_out,
                                  int Nsrc, int M, int Cx, int K1p, int H1, int H2, int H3) {
  extern __shared__ char smem[];
  float*         otile = (float*)smem;                          // 64*16 f32
  _Float16*      msg   = (_Float16*)(smem + 64 * 16 * 4);       // 64*K1p
  _Float16*      a1    = msg + 64 * K1p;                        // 64*H1
  _Float16*      a2    = a1 + 64 * H1;                          // 64*H2
  unsigned char* vfl   = (unsigned char*)(a2 + 64 * H2);        // 64

  int q = blockIdx.x, tid = threadIdx.x;
  int b = q / M;
  float px = pos_dst[(size_t)q * 3], py = pos_dst[(size_t)q * 3 + 1], pz = pos_dst[(size_t)q * 3 + 2];

  if (tid < 64) vfl[tid] = nvalid[(size_t)q * 64 + tid];
  // Build message rows: [x_j (Cx) | pos_j - pos_i (3) | zero pad]
  for (int e = tid; e < 64 * K1p; e += blockDim.x) {
    int r = e / K1p, k = e - r * K1p;
    int j = nidx[(size_t)q * 64 + r];
    j = (j < 0) ? 0 : (j >= Nsrc ? Nsrc - 1 : j);
    float v = 0.0f;
    if (k < Cx) {
      v = x_src[((size_t)b * Nsrc + j) * Cx + k];
    } else if (k < Cx + 3) {
      int c = k - Cx;
      float qc = (c == 0) ? px : (c == 1 ? py : pz);
      v = pos_src[((size_t)b * Nsrc + j) * 3 + c] - qc;
    }
    msg[e] = (_Float16)v;
  }
  __syncthreads();

  int row0 = (tid >> 5) * 16;  // 4 waves, 16 rows each
  wave_layer(msg, K1p, w1, b1, H1, a1, row0, true);
  __syncthreads();
  wave_layer(a1, H1, w2, b2, H2, a2, row0, true);
  __syncthreads();

  int lane = tid & 31, n = lane & 15, mhi = (lane >> 4) * 8;
  for (int n0 = 0; n0 < H3; n0 += 16) {
    v8f acc = {};
#pragma unroll 2
    for (int k0 = 0; k0 < H2; k0 += 32) {
      v16h a = load_a_frag(a2, row0, H2, k0);
      v16h bb = load_b_frag(w3, H3, k0, n0);
      acc = wmma_f16(a, bb, acc);
    }
    __syncthreads();  // prior reducers done with otile
    float bv = b3[n0 + n];
#pragma unroll
    for (int v = 0; v < 8; ++v) {
      int r = row0 + mhi + v;
      float x = acc[v] + bv;
      if (!vfl[r]) x = NEGBIG;
      otile[r * 16 + n] = x;
    }
    __syncthreads();
    if (tid < 16) {
      float mx = -FBIG;
      for (int r = 0; r < 64; ++r) mx = fmaxf(mx, otile[r * 16 + tid]);
      x_out[(size_t)q * H3 + n0 + tid] = fmaxf(mx, 0.0f);  // fused ReLU (+ no-valid -> 0)
    }
  }
}

// ---------------------------------------------------------------------------
// Global MLP: one wave per 16 rows; 288(pad of 259) -> 256 -> 512 -> 1024, WMMA
// ---------------------------------------------------------------------------
__global__ void gmlp_kernel(const float* __restrict__ x2, const float* __restrict__ pos2,
                            const _Float16* __restrict__ w1, const float* __restrict__ b1,
                            const _Float16* __restrict__ w2, const float* __restrict__ b2,
                            const _Float16* __restrict__ w3, const float* __restrict__ b3,
                            float* __restrict__ hout) {
  extern __shared__ char smem[];
  _Float16* in = (_Float16*)smem;      // 16*288
  _Float16* h1 = in + 16 * 288;        // 16*256
  _Float16* h2 = h1 + 16 * 256;        // 16*512
  int row0g = blockIdx.x * 16;
  __builtin_prefetch(w1, 0, 0);

  for (int e = threadIdx.x; e < 16 * 288; e += 32) {
    int r = e / 288, k = e - r * 288;
    size_t gr = (size_t)(row0g + r);
    float v = 0.0f;
    if (k < 256)      v = x2[gr * 256 + k];
    else if (k < 259) v = pos2[gr * 3 + (k - 256)];
    in[e] = (_Float16)v;
  }
  __syncthreads();
  wave_layer(in, 288, w1, b1, 256, h1, 0, true);
  __syncthreads();
  wave_layer(h1, 256, w2, b2, 512, h2, 0, true);
  __syncthreads();

  int lane = threadIdx.x & 31, n = lane & 15, mhi = (lane >> 4) * 8;
  for (int n0 = 0; n0 < 1024; n0 += 16) {
    v8f acc = {};
#pragma unroll 2
    for (int k0 = 0; k0 < 512; k0 += 32) {
      v16h a = load_a_frag(h2, 0, 512, k0);
      v16h bb = load_b_frag(w3, 1024, k0, n0);
      acc = wmma_f16(a, bb, acc);
    }
    float bv = b3[n0 + n];
#pragma unroll
    for (int v = 0; v < 8; ++v)
      hout[(size_t)(row0g + mhi + v) * 1024 + n0 + n] = acc[v] + bv;
  }
}

__global__ void max128_kernel(const float* __restrict__ h, float* __restrict__ out) {
  int g = blockIdx.x >> 2;
  int c = ((blockIdx.x & 3) << 8) + threadIdx.x;
  const float* base = h + (size_t)g * 128 * 1024;
  float m = -FBIG;
  for (int r = 0; r < 128; ++r) m = fmaxf(m, base[(size_t)r * 1024 + c]);
  out[(size_t)g * 1024 + c] = m;
}

__global__ void head_kernel(const float* __restrict__ gmax,
                            const float* __restrict__ l1w, const float* __restrict__ l1b,
                            const float* __restrict__ l2w, const float* __restrict__ l2b,
                            const float* __restrict__ l3w, const float* __restrict__ l3b,
                            float* __restrict__ out7) {
  __shared__ float h1[512];
  __shared__ float h2[256];
  const float* g = gmax + (size_t)blockIdx.x * 1024;
  for (int o = threadIdx.x; o < 512; o += 256) {
    float s = l1b[o];
    for (int k = 0; k < 1024; ++k) s += g[k] * l1w[(size_t)k * 512 + o];
    h1[o] = fmaxf(s, 0.0f);
  }
  __syncthreads();
  for (int o = threadIdx.x; o < 256; o += 256) {
    float s = l2b[o];
    for (int k = 0; k < 512; ++k) s += h1[k] * l2w[(size_t)k * 256 + o];
    h2[o] = fmaxf(s, 0.0f);
  }
  __syncthreads();
  if (threadIdx.x < 7) {
    float s = l3b[threadIdx.x];
    for (int k = 0; k < 256; ++k) s += h2[k] * l3w[(size_t)k * 7 + threadIdx.x];
    out7[(size_t)blockIdx.x * 7 + threadIdx.x] = s;
  }
}

__global__ void final_kernel(const float* __restrict__ out7, float* __restrict__ out) {
  int b = threadIdx.x;
  if (b >= 8) return;
  float v[7];
  for (int c = 0; c < 7; ++c) {
    float s = 0.0f;
    for (int g = 0; g < 4; ++g) s += out7[(size_t)(b * 4 + g) * 7 + c];
    v[c] = s * 0.25f;
  }
  float qn = sqrtf(v[3] * v[3] + v[4] * v[4] + v[5] * v[5] + v[6] * v[6]);
  qn = fmaxf(qn, 1e-12f);
  for (int c = 0; c < 3; ++c) out[b * 7 + c] = v[c];
  for (int c = 3; c < 7; ++c) out[b * 7 + c] = v[c] / qn;
}

// ---------------------------------------------------------------------------
// Host orchestration
// ---------------------------------------------------------------------------
static inline size_t al256(size_t x) { return (x + 255) & ~(size_t)255; }

extern "C" void kernel_launch(void* const* d_in, const int* in_sizes, int n_in,
                              void* d_out, int out_size, void* d_ws, size_t ws_size,
                              hipStream_t stream) {
  (void)in_sizes; (void)n_in; (void)out_size; (void)ws_size;
  const int B = 8, N = 4096, N1 = 2048, N2 = 512, K = 64;

  const float* points = (const float*)d_in[0];
  const float* sa1_w[3] = {(const float*)d_in[1], (const float*)d_in[3], (const float*)d_in[5]};
  const float* sa1_b[3] = {(const float*)d_in[2], (const float*)d_in[4], (const float*)d_in[6]};
  const float* sa2_w[3] = {(const float*)d_in[7], (const float*)d_in[9], (const float*)d_in[11]};
  const float* sa2_b[3] = {(const float*)d_in[8], (const float*)d_in[10], (const float*)d_in[12]};
  const float* g_w[3]   = {(const float*)d_in[13], (const float*)d_in[15], (const float*)d_in[17]};
  const float* g_b[3]   = {(const float*)d_in[14], (const float*)d_in[16], (const float*)d_in[18]};
  const float* l1w = (const float*)d_in[19]; const float* l1b = (const float*)d_in[20];
  const float* l2w = (const float*)d_in[21]; const float* l2b = (const float*)d_in[22];
  const float* l3w = (const float*)d_in[23]; const float* l3b = (const float*)d_in[24];

  // Workspace carve-out
  char* ws = (char*)d_ws; size_t cur = 0;
  auto take = [&](size_t bytes) { char* p = ws + cur; cur = al256(cur + bytes); return p; };

  // f16 weights, K padded to multiple of 32, pre-swizzled to B-fragment order
  const int wK[9]  = {3, 64, 64, 131, 128, 128, 259, 256, 512};
  const int wN[9]  = {64, 64, 128, 128, 128, 256, 256, 512, 1024};
  const int wKp[9] = {32, 64, 64, 160, 128, 128, 288, 256, 512};
  const float* wsrc[9] = {sa1_w[0], sa1_w[1], sa1_w[2], sa2_w[0], sa2_w[1], sa2_w[2],
                          g_w[0], g_w[1], g_w[2]};
  _Float16* wh[9];
  for (int i = 0; i < 9; ++i) wh[i] = (_Float16*)take((size_t)wKp[i] * wN[i] * 2);

  int*   idx1  = (int*)take((size_t)B * N1 * 4);
  float* pos1  = (float*)take((size_t)B * N1 * 3 * 4);
  int*   nidx1 = (int*)take((size_t)B * N1 * K * 4);
  unsigned char* val1 = (unsigned char*)take((size_t)B * N1 * K);
  float* x1    = (float*)take((size_t)B * N1 * 128 * 4);
  int*   idx2  = (int*)take((size_t)B * N2 * 4);
  float* pos2  = (float*)take((size_t)B * N2 * 3 * 4);
  int*   nidx2 = (int*)take((size_t)B * N2 * K * 4);
  unsigned char* val2 = (unsigned char*)take((size_t)B * N2 * K);
  float* x2    = (float*)take((size_t)B * N2 * 256 * 4);
  float* hbuf  = (float*)take((size_t)B * N2 * 1024 * 4);
  float* gmax  = (float*)take((size_t)32 * 1024 * 4);
  float* out7  = (float*)take((size_t)32 * 7 * 4);

  // 1) convert weights to padded, fragment-swizzled f16
  for (int i = 0; i < 9; ++i) {
    int n = wKp[i] * wN[i];
    cvt_w_kernel<<<(n + 255) / 256, 256, 0, stream>>>(wsrc[i], wh[i], wK[i], wN[i], wKp[i]);
  }

  // 2) SA1: FPS -> gather -> radius -> PointConv (WMMA)
  fps_kernel<<<B, 256, 0, stream>>>(points, N, N1, idx1);
  gather_pos_kernel<<<(B * N1 * 3 + 255) / 256, 256, 0, stream>>>(points, idx1, pos1, N, N1, B);
  radius_kernel<<<B * N1, 256, 0, stream>>>(points, pos1, N, N1, 0.01f, nidx1, val1);
  {
    int K1p = 32, H1 = 64, H2 = 64, H3 = 128;
    size_t shm = 64 * 16 * 4 + (size_t)2 * 64 * (K1p + H1 + H2) + 64;
    point_conv_kernel<<<B * N1, 128, shm, stream>>>(
        points, pos1, (const float*)nullptr, nidx1, val1,
        wh[0], sa1_b[0], wh[1], sa1_b[1], wh[2], sa1_b[2],
        x1, N, N1, 0, K1p, H1, H2, H3);
  }

  // 3) SA2
  fps_kernel<<<B, 256, 0, stream>>>(pos1, N1, N2, idx2);
  gather_pos_kernel<<<(B * N2 * 3 + 255) / 256, 256, 0, stream>>>(pos1, idx2, pos2, N1, N2, B);
  radius_kernel<<<B * N2, 256, 0, stream>>>(pos1, pos2, N1, N2, 0.04f, nidx2, val2);
  {
    int K1p = 160, H1 = 128, H2 = 128, H3 = 256;
    size_t shm = 64 * 16 * 4 + (size_t)2 * 64 * (K1p + H1 + H2) + 64;
    point_conv_kernel<<<B * N2, 128, shm, stream>>>(
        pos1, pos2, x1, nidx2, val2,
        wh[3], sa2_b[0], wh[4], sa2_b[1], wh[5], sa2_b[2],
        x2, N1, N2, 128, K1p, H1, H2, H3);
  }

  // 4) global MLP (WMMA) -> max over groups of 128 rows
  {
    size_t shm = (size_t)2 * 16 * (288 + 256 + 512);
    gmlp_kernel<<<(B * N2) / 16, 32, shm, stream>>>(
        x2, pos2, wh[6], g_b[0], wh[7], g_b[1], wh[8], g_b[2], hbuf);
  }
  max128_kernel<<<32 * 4, 256, 0, stream>>>(hbuf, gmax);

  // 5) head + pose output
  head_kernel<<<32, 256, 0, stream>>>(gmax, l1w, l1b, l2w, l2b, l3w, l3b, out7);
  final_kernel<<<1, 8, 0, stream>>>(out7, (float*)d_out);
}
